// BoltzGAT_48765058679017
// MI455X (gfx1250) — compile-verified, matrix operand order checked by reading, not run on hardware
//
#include <hip/hip_runtime.h>

#define NN      100000
#define EE      1600000
#define BB      2048
#define HID     128
#define LAYERS  3
#define HEADS   8
#define HD      16
#define KK      100

typedef __bf16 v16bf __attribute__((ext_vector_type(16)));
typedef float  v8f   __attribute__((ext_vector_type(8)));
typedef unsigned int v4u __attribute__((ext_vector_type(4)));
typedef int          v4i __attribute__((ext_vector_type(4)));
typedef int          v8i __attribute__((ext_vector_type(8)));

#if defined(__has_builtin)
#if __has_builtin(__builtin_amdgcn_tensor_load_to_lds) && \
    __has_builtin(__builtin_amdgcn_s_wait_tensorcnt)
#define HAVE_TDM 1
#endif
#endif
#ifndef HAVE_TDM
#define HAVE_TDM 0
#endif

// ---- ordered-uint encoding for float atomic max (monotonic total order) ----
__device__ __forceinline__ unsigned fenc(float f) {
    unsigned u = __float_as_uint(f);
    return (u & 0x80000000u) ? ~u : (u | 0x80000000u);
}
__device__ __forceinline__ float fdec(unsigned u) {
    unsigned v = (u & 0x80000000u) ? (u & 0x7FFFFFFFu) : ~u;
    return __uint_as_float(v);
}
// init value 0u decodes to NaN -> filtered like reference's isfinite() guard.

// ============================================================================
// K0: convert Wl/Wr (f32, [L][k][n]) -> bf16 stored n-major: wbf[((l*2+m)*128+n)*128+k]
// ============================================================================
__global__ __launch_bounds__(256) void cvt_weights_kernel(
    const float* __restrict__ Wl, const float* __restrict__ Wr,
    __bf16* __restrict__ wbf, int total)
{
    int idx = blockIdx.x * 256 + threadIdx.x;
    if (idx >= total) return;
    int k = idx & 127;
    int n = (idx >> 7) & 127;
    int m = (idx >> 14) & 1;
    int l = idx >> 15;
    const float* W = (m == 0 ? Wl : Wr) + (size_t)l * HID * HID;
    wbf[idx] = (__bf16)W[k * HID + n];
}

// ============================================================================
// K1: xl = h @ Wl_l, xr = h @ Wr_l via v_wmma_f32_16x16x32_bf16.
// Block = 256 threads (8 waves). Block handles a 128-row strip; wave -> 16 rows.
// Weights (both matrices, bf16, n-major) staged into 64KB LDS — via the
// Tensor Data Mover (one DMA descriptor, TENSORcnt-tracked) when available.
// ============================================================================
__global__ __launch_bounds__(256) void gemm_xlxr_kernel(
    const float* __restrict__ h, const __bf16* __restrict__ wbf,
    float* __restrict__ xl, float* __restrict__ xr, int layer, int n_nodes)
{
    __shared__ __bf16 ldsW[2 * HID * HID];   // 65536 bytes, at LDS offset 0

#if HAVE_TDM
    // --- TDM: single descriptor moves 64KB (16384 x 4B words) into LDS ---
    if ((threadIdx.x >> 5) == 0) {           // one wave issues the DMA
        unsigned long long ga =
            (unsigned long long)(const void*)(wbf + (size_t)layer * 2 * HID * HID);
        v4u g0;
        g0[0] = 1u;                                   // count=1, is_restore=0
        g0[1] = 0u;                                   // lds_addr = 0 (ldsW at base)
        g0[2] = (unsigned)(ga & 0xFFFFFFFFull);       // global_addr[31:0]
        g0[3] = (unsigned)((ga >> 32) & 0x01FFFFFFull) | (2u << 30); // addr hi | type=2
        v8i g1;
        g1[0] = 0x00020000;      // workgroup_mask=0, data_size=2 (4B words)
        g1[1] = 0x40000000;      // tensor_dim0 = 16384 (bits 79:48, low half)
        g1[2] = 0x00010000;      // tensor_dim0 hi=0, tensor_dim1 = 1
        g1[3] = 0x40000000;      // tensor_dim1 hi=0, tile_dim0 = 16384
        g1[4] = 0x00000001;      // tile_dim1 = 1, tile_dim2 = 0
        g1[5] = 16384;           // tensor_dim0_stride lo
        g1[6] = 0;               // stride0 hi / tensor_dim1_stride lo
        g1[7] = 0;
        v4i z4 = {0, 0, 0, 0};
#if __clang_major__ >= 23
        v8i z8 = {0, 0, 0, 0, 0, 0, 0, 0};
        __builtin_amdgcn_tensor_load_to_lds(g0, g1, z4, z4, z8, 0);
#else
        __builtin_amdgcn_tensor_load_to_lds(g0, g1, z4, z4, 0);
#endif
        __builtin_amdgcn_s_wait_tensorcnt(0);
    }
    // Escape ldsW into an asm with a memory clobber: the optimizer must now
    // assume the DMA wrote it, so the ds_loads below stay real (provenance is
    // untouched -> still address-space-3 -> still ds_load_b128 selection).
    asm volatile("" : : "v"((const void*)ldsW) : "memory");
    __syncthreads();
#else
    // --- fallback: cooperative 128-bit copy ---
    {
        const uint4* src = (const uint4*)(wbf + (size_t)layer * 2 * HID * HID);
        uint4* dst = (uint4*)ldsW;
        #pragma unroll 4
        for (int i = threadIdx.x; i < (2 * HID * HID * 2) / 16; i += 256)
            dst[i] = src[i];
    }
    __syncthreads();
#endif

    const int wave = threadIdx.x >> 5;      // 0..7
    const int lane = threadIdx.x & 31;
    const int mrow = lane & 15;
    const int half = lane >> 4;
    const int rowBase = blockIdx.x * 128 + wave * 16;

    int rowA = rowBase + mrow;
    if (rowA >= n_nodes) rowA = n_nodes - 1;
    const float* rp = h + (size_t)rowA * HID;

    // A fragments: 16x32 bf16 per 32-wide K chunk (ISA §7.12.2 layout)
    v16bf afrag[4];
    #pragma unroll
    for (int c = 0; c < 4; ++c) {
        #pragma unroll
        for (int p = 0; p < 8; ++p) {
            int kk = 32 * c + ((p < 4) ? 2 * p : 16 + 2 * (p - 4)) + 8 * half;
            float2 f = *(const float2*)(rp + kk);
            afrag[c][2 * p]     = (__bf16)f.x;
            afrag[c][2 * p + 1] = (__bf16)f.y;
        }
    }

    const bool fullStrip = (rowBase + 16) <= n_nodes;   // wave-uniform

    #pragma unroll
    for (int mm = 0; mm < 2; ++mm) {
        float* outp = mm ? xr : xl;
        const __bf16* wmat = ldsW + mm * HID * HID;   // [n][k]
        for (int nt = 0; nt < 8; ++nt) {
            const __bf16* col = wmat + (nt * 16 + mrow) * HID;
            v8f acc = {};
            #pragma unroll
            for (int c = 0; c < 4; ++c) {
                v16bf bfrag;
                #pragma unroll
                for (int p = 0; p < 8; ++p) {
                    int kk = 32 * c + ((p < 4) ? 2 * p : 16 + 2 * (p - 4)) + 8 * half;
                    bfrag[2 * p]     = col[kk];
                    bfrag[2 * p + 1] = col[kk + 1];
                }
                acc = __builtin_amdgcn_wmma_f32_16x16x32_bf16(
                    false, afrag[c], false, bfrag, (short)0, acc, false, false);
            }
            // C/D layout: VGPR r -> row (rowBase + r + 8*half), col = nt*16 + (lane&15)
            if (fullStrip) {
                #pragma unroll
                for (int r = 0; r < 8; ++r)
                    outp[(size_t)(rowBase + r + 8 * half) * HID + nt * 16 + mrow] = acc[r];
            } else {
                #pragma unroll
                for (int r = 0; r < 8; ++r) {
                    int row = rowBase + r + 8 * half;
                    if (row < n_nodes)
                        outp[(size_t)row * HID + nt * 16 + mrow] = acc[r];
                }
            }
        }
    }
}

// ============================================================================
// K2: per-layer init: out accumulator = 0, emax(enc) = 0, denom = 0
// ============================================================================
__global__ __launch_bounds__(256) void init_layer_kernel(
    float* __restrict__ hn, unsigned* __restrict__ emax, float* __restrict__ denom)
{
    int idx = blockIdx.x * 256 + threadIdx.x;
    if (idx < NN * HID) hn[idx] = 0.f;
    if (idx < NN * HEADS) { emax[idx] = 0u; denom[idx] = 0.f; }
}

// ============================================================================
// K3: edge pass 1 — per (edge, head): e = sum_d lrelu(xl[s]+xr[d]) * att;
// store e, atomic-max into emax[dst].
// ============================================================================
__global__ __launch_bounds__(256) void edge_pass1_kernel(
    const int* __restrict__ ei, const float* __restrict__ xl,
    const float* __restrict__ xr, const float* __restrict__ att,
    float* __restrict__ evals, unsigned* __restrict__ emax, int total)
{
    int idx = blockIdx.x * 256 + threadIdx.x;
    if (idx >= total) return;
    int i = idx >> 3, hh = idx & 7;
    int s = ei[i], d = ei[EE + i];
    const float4* pl = (const float4*)(xl + (size_t)s * HID + hh * HD);
    const float4* pr = (const float4*)(xr + (size_t)d * HID + hh * HD);
    const float4* pa = (const float4*)(att + hh * HD);
    float acc = 0.f;
    #pragma unroll
    for (int q = 0; q < 4; ++q) {
        float4 a = pl[q], b = pr[q], w = pa[q];
        float m;
        m = a.x + b.x; acc += (m > 0.f ? m : 0.2f * m) * w.x;
        m = a.y + b.y; acc += (m > 0.f ? m : 0.2f * m) * w.y;
        m = a.z + b.z; acc += (m > 0.f ? m : 0.2f * m) * w.z;
        m = a.w + b.w; acc += (m > 0.f ? m : 0.2f * m) * w.w;
    }
    evals[idx] = acc;
    atomicMax(&emax[(size_t)d * HEADS + hh], fenc(acc));
}

// ============================================================================
// K4: edge pass 2 — ex = exp(e - emax[dst]); denom += ex; hn[dst] += ex*xl[src]
// ============================================================================
__global__ __launch_bounds__(256) void edge_pass2_kernel(
    const int* __restrict__ ei, const float* __restrict__ xl,
    const float* __restrict__ evals, const unsigned* __restrict__ emax,
    float* __restrict__ denom, float* __restrict__ hn, int total)
{
    int idx = blockIdx.x * 256 + threadIdx.x;
    if (idx >= total) return;
    int i = idx >> 3, hh = idx & 7;
    int s = ei[i], d = ei[EE + i];
    const float* lrow = xl + (size_t)s * HID + hh * HD;
    __builtin_prefetch((const void*)lrow, 0, 1);      // global_prefetch: hide gather latency
    float em = fdec(emax[(size_t)d * HEADS + hh]);
    if (!(fabsf(em) < 3.0e38f)) em = 0.f;             // non-finite -> 0 (ref semantics)
    float ex = __expf(evals[idx] - em);
    atomicAdd(&denom[(size_t)d * HEADS + hh], ex);
    const float4* pl = (const float4*)lrow;
    float* o = hn + (size_t)d * HID + hh * HD;
    #pragma unroll
    for (int q = 0; q < 4; ++q) {
        float4 a = pl[q];
        atomicAdd(o + q * 4 + 0, ex * a.x);
        atomicAdd(o + q * 4 + 1, ex * a.y);
        atomicAdd(o + q * 4 + 2, ex * a.z);
        atomicAdd(o + q * 4 + 3, ex * a.w);
    }
}

// ============================================================================
// K5: node finalize: h = elu(acc/denom + bias)
// ============================================================================
__global__ __launch_bounds__(256) void finalize_kernel(
    float* __restrict__ hn, const float* __restrict__ denom,
    const float* __restrict__ bias)
{
    int idx = blockIdx.x * 256 + threadIdx.x;
    if (idx >= NN * HID) return;
    int nnode = idx >> 7, c = idx & 127;
    float dsum = denom[(size_t)nnode * HEADS + (c >> 4)] + 1e-16f;
    float v = hn[idx] / dsum + bias[c];
    hn[idx] = v > 0.f ? v : (__expf(v) - 1.f);
}

// ============================================================================
// Readout
// ============================================================================
__global__ __launch_bounds__(256) void init_pool_kernel(
    float* __restrict__ psum, unsigned* __restrict__ pmaxe)
{
    int idx = blockIdx.x * 256 + threadIdx.x;
    if (idx < BB * HID) { psum[idx] = 0.f; pmaxe[idx] = 0u; }
}

__global__ __launch_bounds__(256) void gate_kernel(
    const float* __restrict__ h, const float* __restrict__ Wg,
    const float* __restrict__ bg, float* __restrict__ gate)
{
    int i = blockIdx.x * 256 + threadIdx.x;
    if (i >= NN) return;
    const float* r = h + (size_t)i * HID;
    float z = bg[0];
    #pragma unroll 4
    for (int c = 0; c < HID; ++c) z += r[c] * Wg[c];
    gate[i] = 1.f / (1.f + __expf(-z));
}

__global__ __launch_bounds__(256) void pool_kernel(
    const float* __restrict__ h, const float* __restrict__ gate,
    const int* __restrict__ batch, float* __restrict__ psum,
    unsigned* __restrict__ pmaxe)
{
    int idx = blockIdx.x * 256 + threadIdx.x;
    if (idx >= NN * HID) return;
    int nnode = idx >> 7, c = idx & 127;
    int b = batch[nnode];
    float v = h[idx];
    atomicAdd(&psum[(size_t)b * HID + c], gate[nnode] * v);
    atomicMax(&pmaxe[(size_t)b * HID + c], fenc(v));
}

// One block (128 threads) per graph: energies, temperature softmax, MLP.
__global__ __launch_bounds__(128) void head_kernel(
    const float* __restrict__ psum, const unsigned* __restrict__ pmaxe,
    const float* __restrict__ temps, const float* __restrict__ We,
    const float* __restrict__ be, const float* __restrict__ W1,
    const float* __restrict__ b1, const float* __restrict__ W2,
    const float* __restrict__ b2, float* __restrict__ out)
{
    __shared__ float mol[2 * HID];
    __shared__ float prob[KK];
    __shared__ float h1[64];
    __shared__ float sred[2];
    int b = blockIdx.x, t = threadIdx.x;

    float mv = fdec(pmaxe[(size_t)b * HID + t]);
    if (!(fabsf(mv) < 3.0e38f)) mv = 0.f;
    mol[t]       = psum[(size_t)b * HID + t];
    mol[HID + t] = mv;
    __syncthreads();

    float invT = 1.f / temps[b];
    if (t < KK) {
        float acc = be[t];
        for (int j = 0; j < 2 * HID; ++j) acc += mol[j] * We[j * KK + t];
        prob[t] = -acc * invT;
    }
    __syncthreads();
    if (t == 0) {
        float mx = prob[0];
        for (int k = 1; k < KK; ++k) mx = fmaxf(mx, prob[k]);
        sred[0] = mx;
    }
    __syncthreads();
    if (t < KK) prob[t] = __expf(prob[t] - sred[0]);
    __syncthreads();
    if (t == 0) {
        float s = 0.f;
        for (int k = 0; k < KK; ++k) s += prob[k];
        sred[1] = s;
    }
    __syncthreads();
    if (t < KK) prob[t] /= sred[1];
    __syncthreads();
    if (t < 64) {
        float acc = b1[t];
        for (int k = 0; k < KK; ++k) acc += prob[k] * W1[k * 64 + t];
        h1[t] = acc > 0.f ? acc : (__expf(acc) - 1.f);
    }
    __syncthreads();
    if (t == 0) {
        float o = b2[0];
        for (int j = 0; j < 64; ++j) o += h1[j] * W2[j];
        out[b] = o;
    }
}

// ============================================================================
extern "C" void kernel_launch(void* const* d_in, const int* in_sizes, int n_in,
                              void* d_out, int out_size, void* d_ws, size_t ws_size,
                              hipStream_t stream)
{
    const float* x     = (const float*)d_in[0];
    const int*   ei    = (const int*)d_in[1];
    const int*   batch = (const int*)d_in[2];
    const float* temps = (const float*)d_in[3];
    const float* Wl    = (const float*)d_in[4];
    const float* Wr    = (const float*)d_in[5];
    const float* att   = (const float*)d_in[6];
    const float* bias  = (const float*)d_in[7];
    const float* Wg    = (const float*)d_in[8];
    const float* bg    = (const float*)d_in[9];
    const float* We    = (const float*)d_in[10];
    const float* be    = (const float*)d_in[11];
    const float* W1    = (const float*)d_in[12];
    const float* b1    = (const float*)d_in[13];
    const float* W2    = (const float*)d_in[14];
    const float* b2    = (const float*)d_in[15];
    float* out = (float*)d_out;

    // ---- workspace carve-up (256B aligned slices) ----
    char* cur = (char*)d_ws;
    auto take = [&](size_t bytes) -> void* {
        void* p = (void*)cur;
        cur += (bytes + 255) & ~(size_t)255;
        return p;
    };
    float*    bufA  = (float*)take((size_t)NN * HID * 4);
    float*    bufB  = (float*)take((size_t)NN * HID * 4);
    float*    xl    = (float*)take((size_t)NN * HID * 4);
    float*    xr    = (float*)take((size_t)NN * HID * 4);
    float*    evals = (float*)take((size_t)EE * HEADS * 4);
    unsigned* emax  = (unsigned*)take((size_t)NN * HEADS * 4);
    float*    denom = (float*)take((size_t)NN * HEADS * 4);
    float*    gate  = (float*)take((size_t)NN * 4);
    float*    psum  = (float*)take((size_t)BB * HID * 4);
    unsigned* pmaxe = (unsigned*)take((size_t)BB * HID * 4);
    __bf16*   wbf   = (__bf16*)take((size_t)LAYERS * 2 * HID * HID * 2);

    const int nodeElems  = NN * HID;                 // 12.8M
    const int nodeBlocks = (nodeElems + 255) / 256;  // 50000
    const int edgeElems  = EE * HEADS;               // 12.8M
    const int edgeBlocks = (edgeElems + 255) / 256;  // 50000
    const int gemmBlocks = (NN + 127) / 128;         // 782

    // weight conversion (once per launch; deterministic)
    {
        int total = LAYERS * 2 * HID * HID;
        cvt_weights_kernel<<<(total + 255) / 256, 256, 0, stream>>>(Wl, Wr, wbf, total);
    }

    // ---- GATv2 layers ----
    const float* hcur = x;
    float* hn = bufA;
    for (int l = 0; l < LAYERS; ++l) {
        hn = (l == 1) ? bufB : bufA;
        gemm_xlxr_kernel<<<gemmBlocks, 256, 0, stream>>>(hcur, wbf, xl, xr, l, NN);
        init_layer_kernel<<<nodeBlocks, 256, 0, stream>>>(hn, emax, denom);
        edge_pass1_kernel<<<edgeBlocks, 256, 0, stream>>>(ei, xl, xr, att + l * HID,
                                                          evals, emax, edgeElems);
        edge_pass2_kernel<<<edgeBlocks, 256, 0, stream>>>(ei, xl, evals, emax,
                                                          denom, hn, edgeElems);
        finalize_kernel<<<nodeBlocks, 256, 0, stream>>>(hn, denom, bias + l * HID);
        hcur = hn;
    }

    // ---- readout ----
    init_pool_kernel<<<(BB * HID + 255) / 256, 256, 0, stream>>>(psum, pmaxe);
    gate_kernel<<<(NN + 255) / 256, 256, 0, stream>>>(hcur, Wg, bg, gate);
    pool_kernel<<<nodeBlocks, 256, 0, stream>>>(hcur, gate, batch, psum, pmaxe);
    head_kernel<<<BB, 128, 0, stream>>>(psum, pmaxe, temps, We, be, W1, b1, W2, b2, out);
}
